// Block_47794396070541
// MI455X (gfx1250) — matrix operand order repaired and checked by reading, not compiled, not run on hardware
//
#include <hip/hip_runtime.h>
#include <cstdint>

typedef __attribute__((ext_vector_type(16))) _Float16 v16h;
typedef __attribute__((ext_vector_type(8)))  _Float16 v8h;
typedef __attribute__((ext_vector_type(8)))  float    v8f;
typedef __attribute__((ext_vector_type(4)))  unsigned int u32x4;
typedef __attribute__((ext_vector_type(8)))  int      i32x8;
typedef __attribute__((ext_vector_type(4)))  int      i32x4;

union U16h { v16h v; v8h h[2]; };

// ---------------- problem constants ----------------
#define BB 2
#define SS 2048
#define DD 1024
#define NHH 16
#define MHH 4
#define HDD 64
#define GG 12
#define EE 8
#define II 1024
#define ISHH 1024
#define TT (BB*SS)   // 4096 tokens

#if defined(__HIP_DEVICE_COMPILE__) && __has_builtin(__builtin_amdgcn_tensor_load_to_lds)
#define USE_TDM 1
#else
#define USE_TDM 0
#endif

#if USE_TDM
// Issue a 2-D TDM load: rows x rowlen (f16 elements), row stride = stride_elts,
// contiguous destination tile in LDS at byte offset lds_off.
__device__ __forceinline__ void tdm_load_2d(unsigned lds_off, const void* gptr,
                                            unsigned rows, unsigned rowlen,
                                            unsigned stride_elts) {
  unsigned long long ga = (unsigned long long)gptr;
  unsigned td0 = stride_elts;       // tensor_dim0 bound (>= rowlen)
  unsigned td1 = 1u << 20;          // generous row bound
  u32x4 g0;
  g0[0] = 1u;                                        // count=1, user descriptor
  g0[1] = lds_off;                                   // lds_addr
  g0[2] = (unsigned)(ga & 0xffffffffu);              // global_addr[31:0]
  g0[3] = (unsigned)((ga >> 32) & 0x01ffffffu) | (2u << 30);  // addr[56:32] | type=2
  i32x8 g1;
  g1[0] = (int)(1u << 16);                           // data_size=1 (2 bytes)
  g1[1] = (int)((td0 & 0xffffu) << 16);              // tensor_dim0[15:0]
  g1[2] = (int)(((td0 >> 16) & 0xffffu) | ((td1 & 0xffffu) << 16));
  g1[3] = (int)(((td1 >> 16) & 0xffffu) | ((rowlen & 0xffffu) << 16)); // tile_dim0
  g1[4] = (int)(rows & 0xffffu);                     // tile_dim1 (tile_dim2=0)
  g1[5] = (int)stride_elts;                          // tensor_dim0_stride[31:0]
  g1[6] = 0;
  g1[7] = 0;
  i32x4 gz = {0, 0, 0, 0};
#if __clang_major__ >= 23
  i32x8 gz8 = {0, 0, 0, 0, 0, 0, 0, 0};
  __builtin_amdgcn_tensor_load_to_lds(g0, g1, gz, gz, gz8, 0);
#else
  __builtin_amdgcn_tensor_load_to_lds(g0, g1, gz, gz, 0);
#endif
}
#endif

// =====================================================================
// RMSNorm: f32 in -> f16 out (one block per token)
// =====================================================================
__global__ __launch_bounds__(256) void rmsnorm_f16(const float* __restrict__ x,
                                                   const float* __restrict__ w,
                                                   _Float16* __restrict__ out) {
  int t = blockIdx.x;
  const float* xr = x + (size_t)t * DD;
  float s = 0.f;
  for (int d = threadIdx.x; d < DD; d += 256) { float v = xr[d]; s += v * v; }
  for (int m = 16; m >= 1; m >>= 1) s += __shfl_xor(s, m, 32);
  __shared__ float red[8];
  __shared__ float rnorm;
  if ((threadIdx.x & 31) == 0) red[threadIdx.x >> 5] = s;
  __syncthreads();
  if (threadIdx.x == 0) {
    float tot = 0.f;
    for (int i = 0; i < 8; i++) tot += red[i];
    rnorm = rsqrtf(tot * (1.0f / DD) + 1e-5f);
  }
  __syncthreads();
  float r = rnorm;
  for (int d = threadIdx.x; d < DD; d += 256)
    out[(size_t)t * DD + d] = (_Float16)(xr[d] * r * w[d]);
}

// =====================================================================
// Weight convert+transpose: in f32 [K][N] -> out f16 [N][K]
// =====================================================================
__global__ void transpose_to_f16(const float* __restrict__ in,
                                 _Float16* __restrict__ out, int K, int N) {
  long idx = (long)blockIdx.x * 256 + threadIdx.x;
  if (idx >= (long)K * N) return;
  int k = (int)(idx / N), n = (int)(idx % N);
  out[(size_t)n * K + k] = (_Float16)in[idx];
}

// =====================================================================
// WMMA GEMM: C[M][N] f32 = A[M][K] f16 (row-major) * Bt[N][K] f16
// 128 thr (4 waves); block tile 128x128; wave tile 64x64 (16 WMMA / K-step,
// each A/B fragment reused 4x); BK=32; double-buffered LDS fed by the TDM
// (tensor_load_to_lds + s_wait_tensorcnt) when available.
// =====================================================================
__global__ __launch_bounds__(128) void gemm_f16_wmma(const _Float16* __restrict__ A,
                                                     const _Float16* __restrict__ Bt,
                                                     float* __restrict__ C,
                                                     int M, int N, int Kd) {
  constexpr int BK = 32;
  constexpr int TILE = 128 * BK;           // halves per operand tile
  __shared__ _Float16 As[2][TILE];
  __shared__ _Float16 Bs[2][TILE];
  int tid = threadIdx.x;
  int wave = tid >> 5, lane = tid & 31;
  int lh = lane & 15, hi = lane >> 4;
  int nBlkN = N >> 7;
  int bm = (blockIdx.x / nBlkN) << 7;
  int bn = (blockIdx.x % nBlkN) << 7;
  int wm = (wave & 1) * 64;
  int wn = (wave >> 1) * 64;
  int nk = Kd / BK;

  v8f acc[4][4] = {};

#if USE_TDM
  unsigned ldsA = (unsigned)(uintptr_t)(&As[0][0]);
  unsigned ldsB = (unsigned)(uintptr_t)(&Bs[0][0]);
  bool issuer = (wave == 0);
  if (issuer) {
    tdm_load_2d(ldsA, &A[(size_t)bm * Kd], 128, BK, (unsigned)Kd);
    tdm_load_2d(ldsB, &Bt[(size_t)bn * Kd], 128, BK, (unsigned)Kd);
  }
  for (int k = 0; k < nk; k++) {
    int buf = k & 1;
    if (issuer) {
      if (k + 1 < nk) {
        unsigned nb = (unsigned)((k + 1) & 1) * (TILE * 2u);
        tdm_load_2d(ldsA + nb, &A[(size_t)bm * Kd + (size_t)(k + 1) * BK], 128, BK, (unsigned)Kd);
        tdm_load_2d(ldsB + nb, &Bt[(size_t)bn * Kd + (size_t)(k + 1) * BK], 128, BK, (unsigned)Kd);
        __builtin_amdgcn_s_wait_tensorcnt(2);   // tile k landed (in-order TDM)
      } else {
        __builtin_amdgcn_s_wait_tensorcnt(0);
      }
    }
    __syncthreads();
    U16h a[4];
#pragma unroll
    for (int i = 0; i < 4; i++) {
      const _Float16* ap = &As[buf][(wm + i * 16 + lh) * BK + hi * 8];
      a[i].h[0] = *(const v8h*)(ap);
      a[i].h[1] = *(const v8h*)(ap + 16);
    }
#pragma unroll
    for (int g = 0; g < 4; g++) {
      U16h b;
      const _Float16* bp = &Bs[buf][(wn + g * 16 + lh) * BK + hi * 8];
      b.h[0] = *(const v8h*)(bp);
      b.h[1] = *(const v8h*)(bp + 16);
#pragma unroll
      for (int mi = 0; mi < 4; mi++)
        acc[mi][g] = __builtin_amdgcn_wmma_f32_16x16x32_f16(false, a[mi].v, false, b.v, (short)0, acc[mi][g], false, false);
    }
    __syncthreads();
  }
#else
  // Fallback: double-buffered register staging (each thread moves one 64B row
  // of A and one of B per tile)
  float4 ra[4], rb[4];
  auto stageRegs = [&](int k) {
    const float4* pa = (const float4*)&A[(size_t)(bm + tid) * Kd + (size_t)k * BK];
    const float4* pb = (const float4*)&Bt[(size_t)(bn + tid) * Kd + (size_t)k * BK];
#pragma unroll
    for (int i = 0; i < 4; i++) { ra[i] = pa[i]; rb[i] = pb[i]; }
  };
  auto stageStore = [&](int buf) {
    float4* da = (float4*)&As[buf][tid * BK];
    float4* db = (float4*)&Bs[buf][tid * BK];
#pragma unroll
    for (int i = 0; i < 4; i++) { da[i] = ra[i]; db[i] = rb[i]; }
  };
  stageRegs(0);
  stageStore(0);
  for (int k = 0; k < nk; k++) {
    __syncthreads();
    if (k + 1 < nk) stageRegs(k + 1);
    int buf = k & 1;
    U16h a[4];
#pragma unroll
    for (int i = 0; i < 4; i++) {
      const _Float16* ap = &As[buf][(wm + i * 16 + lh) * BK + hi * 8];
      a[i].h[0] = *(const v8h*)(ap);
      a[i].h[1] = *(const v8h*)(ap + 16);
    }
#pragma unroll
    for (int g = 0; g < 4; g++) {
      U16h b;
      const _Float16* bp = &Bs[buf][(wn + g * 16 + lh) * BK + hi * 8];
      b.h[0] = *(const v8h*)(bp);
      b.h[1] = *(const v8h*)(bp + 16);
#pragma unroll
      for (int mi = 0; mi < 4; mi++)
        acc[mi][g] = __builtin_amdgcn_wmma_f32_16x16x32_f16(false, a[mi].v, false, b.v, (short)0, acc[mi][g], false, false);
    }
    __syncthreads();
    if (k + 1 < nk) stageStore((k + 1) & 1);
  }
#endif

#pragma unroll
  for (int mi = 0; mi < 4; mi++)
#pragma unroll
    for (int g = 0; g < 4; g++)
#pragma unroll
      for (int r = 0; r < 8; r++) {
        size_t gm = (size_t)(bm + wm + mi * 16 + r + hi * 8);
        size_t gn = (size_t)(bn + wn + g * 16 + lh);
        C[gm * N + gn] = acc[mi][g][r];
      }
}

// =====================================================================
// QK-norm + RoPE.  in: [t][nheads*64] f32. out: [b][head][s][d] f16
// one wave per (token, head)
// =====================================================================
__global__ __launch_bounds__(256) void rope_norm(const float* __restrict__ in,
                                                 _Float16* __restrict__ out, int nheads) {
  int wave = threadIdx.x >> 5, lane = threadIdx.x & 31;
  int wg = blockIdx.x * 8 + wave;
  int t = wg / nheads, hh = wg % nheads;
  int s = t & (SS - 1), b = t >> 11;
  const float* xr = in + (size_t)t * (nheads * HDD) + hh * HDD;
  float v0 = xr[lane], v1 = xr[lane + 32];
  float ss = v0 * v0 + v1 * v1;
  for (int m = 16; m >= 1; m >>= 1) ss += __shfl_xor(ss, m, 32);
  float rn = rsqrtf(ss * (1.0f / HDD) + 1e-6f);
  v0 *= rn; v1 *= rn;
  int j = lane & 15;
  float ang = (float)s * exp2f(-10.0f * (float)j * (1.0f / 16.0f));  // 1024^{-j/16}
  float c = cosf(ang), sn = sinf(ang);
  float part = __shfl_xor(v0, 16, 32);
  float r0 = (lane < 16) ? (v0 * c - part * sn) : (v0 * c + part * sn);
  _Float16* ob = out + (((size_t)(b * nheads + hh)) * SS + s) * HDD;
  ob[lane]      = (_Float16)r0;
  ob[lane + 32] = (_Float16)v1;
}

// V f32 [t][MH*HD] -> f16 transposed [b][kvh][d][s]
__global__ void transpose_v(const float* __restrict__ v, _Float16* __restrict__ vt) {
  long idx = (long)blockIdx.x * 256 + threadIdx.x;   // over TT*MHH*HDD
  int t = (int)(idx >> 8);
  int r = (int)(idx & 255);
  int kvh = r >> 6, d = r & 63;
  int b = t >> 11, s = t & (SS - 1);
  vt[(((size_t)(b * MHH + kvh)) * HDD + d) * SS + s] = (_Float16)v[idx];
}

// attention output gate: gate[t][h] = 2*sigmoid(xn1[t,:G] @ ag)
__global__ void attn_gate_kernel(const _Float16* __restrict__ xn1,
                                 const float* __restrict__ ag,
                                 float* __restrict__ gate) {
  int idx = blockIdx.x * 256 + threadIdx.x;
  int t = idx >> 4, h = idx & 15;
  float acc = 0.f;
#pragma unroll
  for (int g = 0; g < GG; g++) acc += (float)xn1[(size_t)t * DD + g] * ag[g * NHH + h];
  gate[idx] = 2.0f / (1.0f + __expf(-acc));
}

// =====================================================================
// Flash attention: one wave per (b, head, 16-query tile), causal.
// Scores and PV via WMMA; softmax via wave32 shuffles; P via per-wave LDS.
// =====================================================================
__global__ __launch_bounds__(256) void flash_attn(const _Float16* __restrict__ q16,
                                                  const _Float16* __restrict__ k16,
                                                  const _Float16* __restrict__ vt16,
                                                  const float* __restrict__ gate,
                                                  _Float16* __restrict__ out) {
  constexpr int QT = 16, KT = 32;
  __shared__ _Float16 pls[8][QT * KT];
  int wave = threadIdx.x >> 5, lane = threadIdx.x & 31;
  int lh = lane & 15, hi = lane >> 4;
  int wg = blockIdx.x * 8 + wave;
  int qt = wg & 127;
  int h  = (wg >> 7) & 15;
  int b  = wg >> 11;
  int kvh = h >> 2;

  const _Float16* qbase = q16 + (((size_t)(b * NHH + h)) * SS + qt * QT) * HDD;
  const _Float16* kbase = k16 + ((size_t)(b * MHH + kvh)) * SS * HDD;
  const _Float16* vbase = vt16 + ((size_t)(b * MHH + kvh)) * HDD * SS;

  U16h qa[2];
#pragma unroll
  for (int c = 0; c < 2; c++) {
    const _Float16* qp = qbase + (size_t)lh * HDD + c * 32 + hi * 8;
    qa[c].h[0] = *(const v8h*)(qp);
    qa[c].h[1] = *(const v8h*)(qp + 16);
  }

  float rowM[8], rowS[8];
  v8f o[4] = {};
#pragma unroll
  for (int r = 0; r < 8; r++) { rowM[r] = -1e30f; rowS[r] = 0.f; }

  int nkt = (qt >> 1) + 1;
  for (int kt = 0; kt < nkt; kt++) {
    v8f sc0 = {}; v8f sc1 = {};
#pragma unroll
    for (int c = 0; c < 2; c++) {
      U16h kb;
      const _Float16* kp = kbase + (size_t)(kt * KT + lh) * HDD + c * 32 + hi * 8;
      kb.h[0] = *(const v8h*)(kp);
      kb.h[1] = *(const v8h*)(kp + 16);
      sc0 = __builtin_amdgcn_wmma_f32_16x16x32_f16(false, qa[c].v, false, kb.v, (short)0, sc0, false, false);
    }
#pragma unroll
    for (int c = 0; c < 2; c++) {
      U16h kb;
      const _Float16* kp = kbase + (size_t)(kt * KT + 16 + lh) * HDD + c * 32 + hi * 8;
      kb.h[0] = *(const v8h*)(kp);
      kb.h[1] = *(const v8h*)(kp + 16);
      sc1 = __builtin_amdgcn_wmma_f32_16x16x32_f16(false, qa[c].v, false, kb.v, (short)0, sc1, false, false);
    }
#pragma unroll
    for (int r = 0; r < 8; r++) {
      int row = qt * QT + r + hi * 8;
      int key0 = kt * KT + lh;
      int key1 = key0 + 16;
      float v0 = (key0 > row) ? -1e30f : sc0[r] * 0.125f;   // 1/sqrt(64)
      float v1 = (key1 > row) ? -1e30f : sc1[r] * 0.125f;
      float m = fmaxf(v0, v1);
      for (int msk = 8; msk >= 1; msk >>= 1) m = fmaxf(m, __shfl_xor(m, msk, 32));
      float mnew = fmaxf(rowM[r], m);
      float sc_old = __expf(rowM[r] - mnew);
      rowM[r] = mnew;
      float p0 = (key0 > row) ? 0.f : __expf(v0 - mnew);
      float p1 = (key1 > row) ? 0.f : __expf(v1 - mnew);
      float ps = p0 + p1;
      for (int msk = 8; msk >= 1; msk >>= 1) ps += __shfl_xor(ps, msk, 32);
      rowS[r] = rowS[r] * sc_old + ps;
#pragma unroll
      for (int j = 0; j < 4; j++) o[j][r] *= sc_old;
      pls[wave][(r + hi * 8) * KT + lh]      = (_Float16)p0;
      pls[wave][(r + hi * 8) * KT + 16 + lh] = (_Float16)p1;
    }
    asm volatile("s_wait_dscnt 0" ::: "memory");   // wave-local LDS RAW fence
    U16h pa;
    {
      const _Float16* pp = &pls[wave][lh * KT + hi * 8];
      pa.h[0] = *(const v8h*)(pp);
      pa.h[1] = *(const v8h*)(pp + 16);
    }
#pragma unroll
    for (int j = 0; j < 4; j++) {
      U16h vb;
      const _Float16* vp = vbase + (size_t)(j * 16 + lh) * SS + kt * KT + hi * 8;
      vb.h[0] = *(const v8h*)(vp);
      vb.h[1] = *(const v8h*)(vp + 16);
      o[j] = __builtin_amdgcn_wmma_f32_16x16x32_f16(false, pa.v, false, vb.v, (short)0, o[j], false, false);
    }
  }
#pragma unroll
  for (int r = 0; r < 8; r++) {
    int row = qt * QT + r + hi * 8;
    int t = b * SS + row;
    float g = gate[(size_t)t * NHH + h];
    float inv = (rowS[r] > 0.f) ? (1.0f / rowS[r]) : 0.f;
#pragma unroll
    for (int j = 0; j < 4; j++)
      out[(size_t)t * (NHH * HDD) + h * HDD + j * 16 + lh] = (_Float16)(o[j][r] * inv * g);
  }
}

// =====================================================================
// Router top-2 (softmax denominators cancel -> logistic of logit diff)
// =====================================================================
__global__ __launch_bounds__(256) void router_topk(const _Float16* __restrict__ xn2,
                                                   const float* __restrict__ router,
                                                   float* __restrict__ combine) {
  int wave = threadIdx.x >> 5, lane = threadIdx.x & 31;
  int t = blockIdx.x * 8 + wave;
  float acc[EE];
#pragma unroll
  for (int e = 0; e < EE; e++) acc[e] = 0.f;
  for (int d = lane; d < DD; d += 32) {
    float xv = (float)xn2[(size_t)t * DD + d];
    const float* rr = router + (size_t)d * EE;
#pragma unroll
    for (int e = 0; e < EE; e++) acc[e] += xv * rr[e];
  }
#pragma unroll
  for (int e = 0; e < EE; e++)
    for (int m = 16; m >= 1; m >>= 1) acc[e] += __shfl_xor(acc[e], m, 32);
  if (lane == 0) {
    int a = 0;
    for (int e = 1; e < EE; e++) if (acc[e] > acc[a]) a = e;
    int b2 = (a == 0) ? 1 : 0;
    for (int e = 0; e < EE; e++) if (e != a && acc[e] > acc[b2]) b2 = e;
    float wa = 1.0f / (1.0f + __expf(acc[b2] - acc[a]));
    float wb = 1.0f - wa;
    float* cb = combine + (size_t)t * EE;
    for (int e = 0; e < EE; e++) cb[e] = (e == a) ? wa : ((e == b2) ? wb : 0.f);
  }
}

// act = silu(G) * U  -> f16, with row strides (handles fused [gate|up] layout)
__global__ void silu_mul(const float* __restrict__ G, long gs,
                         const float* __restrict__ U, long us,
                         _Float16* __restrict__ out, int cols, long n) {
  long idx = (long)blockIdx.x * 256 + threadIdx.x;
  if (idx >= n) return;
  long t = idx / cols; int i = (int)(idx % cols);
  float g = G[t * gs + i];
  float u = U[t * us + i];
  out[idx] = (_Float16)((g / (1.0f + __expf(-g))) * u);
}

__global__ void zero_f32(float* __restrict__ p, long n) {
  long idx = (long)blockIdx.x * 256 + threadIdx.x;
  if (idx < n) p[idx] = 0.f;
}

__global__ void moe_accum(const float* __restrict__ y, const float* __restrict__ combine,
                          int e, float* __restrict__ acc) {
  long idx = (long)blockIdx.x * 256 + threadIdx.x;
  long t = idx >> 10;   // / DD
  acc[idx] += combine[t * EE + e] * y[idx];
}

__global__ void add2(const float* __restrict__ a, const float* __restrict__ b,
                     float* __restrict__ c) {
  long idx = (long)blockIdx.x * 256 + threadIdx.x;
  c[idx] = a[idx] + b[idx];
}

__global__ void add3(const float* __restrict__ a, const float* __restrict__ b,
                     const float* __restrict__ c, float* __restrict__ d) {
  long idx = (long)blockIdx.x * 256 + threadIdx.x;
  d[idx] = a[idx] + b[idx] + c[idx];
}

// =====================================================================
extern "C" void kernel_launch(void* const* d_in, const int* in_sizes, int n_in,
                              void* d_out, int out_size, void* d_ws, size_t ws_size,
                              hipStream_t stream) {
  const float* x        = (const float*)d_in[0];
  const float* ln1      = (const float*)d_in[1];
  const float* ln2      = (const float*)d_in[2];
  const float* w_q      = (const float*)d_in[3];
  const float* w_k      = (const float*)d_in[4];
  const float* w_v      = (const float*)d_in[5];
  const float* w_o      = (const float*)d_in[6];
  const float* ag       = (const float*)d_in[7];
  const float* router   = (const float*)d_in[8];
  const float* w_ug     = (const float*)d_in[9];
  const float* w_dm     = (const float*)d_in[10];
  const float* w_gsh    = (const float*)d_in[11];
  const float* w_ush    = (const float*)d_in[12];
  const float* w_dsh    = (const float*)d_in[13];
  float* out = (float*)d_out;

  char* p = (char*)d_ws;
  auto aF = [&](size_t n) { float* r = (float*)p; p += n * 4; p = (char*)(((uintptr_t)p + 255) & ~(uintptr_t)255); return r; };
  auto aH = [&](size_t n) { _Float16* r = (_Float16*)p; p += n * 2; p = (char*)(((uintptr_t)p + 255) & ~(uintptr_t)255); return r; };

  _Float16* xn1    = aH((size_t)TT * DD);
  _Float16* wqT    = aH((size_t)DD * DD);
  _Float16* wkT    = aH((size_t)DD * MHH * HDD);
  _Float16* wvT    = aH((size_t)DD * MHH * HDD);
  _Float16* woT    = aH((size_t)DD * DD);
  float*    qf     = aF((size_t)TT * DD);
  float*    kf     = aF((size_t)TT * MHH * HDD);
  float*    vf     = aF((size_t)TT * MHH * HDD);
  _Float16* q16    = aH((size_t)TT * DD);
  _Float16* k16    = aH((size_t)TT * MHH * HDD);
  _Float16* vt16   = aH((size_t)TT * MHH * HDD);
  float*    gateb  = aF((size_t)TT * NHH);
  _Float16* attn16 = aH((size_t)TT * DD);
  float*    aproj  = aF((size_t)TT * DD);
  float*    hbuf   = aF((size_t)TT * DD);
  _Float16* xn2    = aH((size_t)TT * DD);
  float*    comb   = aF((size_t)TT * EE);
  _Float16* weT    = aH((size_t)2 * II * DD);     // per-expert up_gate^T (2048x1024)
  _Float16* wdT    = aH((size_t)DD * II);         // per-expert down^T
  float*    ugbuf  = aF((size_t)TT * 2 * II);
  _Float16* act16  = aH((size_t)TT * II);
  float*    ybuf   = aF((size_t)TT * DD);
  float*    moeo   = aF((size_t)TT * DD);
  _Float16* wgT    = aH((size_t)ISHH * DD);
  _Float16* wuT    = aH((size_t)ISHH * DD);
  _Float16* wdshT  = aH((size_t)DD * ISHH);
  float*    gsh    = aF((size_t)TT * ISHH);
  float*    ush    = aF((size_t)TT * ISHH);
  _Float16* actsh  = aH((size_t)TT * ISHH);
  float*    shout  = aF((size_t)TT * DD);

  const long ND = (long)TT * DD;
  auto gemm = [&](const _Float16* A, const _Float16* Bt, float* C, int M, int N, int K) {
    gemm_f16_wmma<<<dim3((M / 128) * (N / 128)), dim3(128), 0, stream>>>(A, Bt, C, M, N, K);
  };
  auto tr = [&](const float* in, _Float16* o, int K, int N) {
    long n = (long)K * N;
    transpose_to_f16<<<dim3((unsigned)((n + 255) / 256)), dim3(256), 0, stream>>>(in, o, K, N);
  };

  // ---- attention ----
  rmsnorm_f16<<<dim3(TT), dim3(256), 0, stream>>>(x, ln1, xn1);
  tr(w_q, wqT, DD, DD);
  tr(w_k, wkT, DD, MHH * HDD);
  tr(w_v, wvT, DD, MHH * HDD);
  tr(w_o, woT, DD, DD);
  gemm(xn1, wqT, qf, TT, DD, DD);
  gemm(xn1, wkT, kf, TT, MHH * HDD, DD);
  gemm(xn1, wvT, vf, TT, MHH * HDD, DD);
  rope_norm<<<dim3(TT * NHH / 8), dim3(256), 0, stream>>>(qf, q16, NHH);
  rope_norm<<<dim3(TT * MHH / 8), dim3(256), 0, stream>>>(kf, k16, MHH);
  transpose_v<<<dim3(TT * MHH * HDD / 256), dim3(256), 0, stream>>>(vf, vt16);
  attn_gate_kernel<<<dim3(TT * NHH / 256), dim3(256), 0, stream>>>(xn1, ag, gateb);
  flash_attn<<<dim3(BB * NHH * (SS / 16) / 8), dim3(256), 0, stream>>>(q16, k16, vt16, gateb, attn16);
  gemm(attn16, woT, aproj, TT, DD, DD);
  add2<<<dim3((unsigned)(ND / 256)), dim3(256), 0, stream>>>(x, aproj, hbuf);

  // ---- second norm + router ----
  rmsnorm_f16<<<dim3(TT), dim3(256), 0, stream>>>(hbuf, ln2, xn2);
  router_topk<<<dim3(TT / 8), dim3(256), 0, stream>>>(xn2, router, comb);

  // ---- MoE (dense per expert, weighted combine) ----
  zero_f32<<<dim3((unsigned)(ND / 256)), dim3(256), 0, stream>>>(moeo, ND);
  for (int e = 0; e < EE; e++) {
    tr(w_ug + (size_t)e * DD * 2 * II, weT, DD, 2 * II);
    gemm(xn2, weT, ugbuf, TT, 2 * II, DD);
    silu_mul<<<dim3((unsigned)(((long)TT * II + 255) / 256)), dim3(256), 0, stream>>>(
        ugbuf, 2 * II, ugbuf + II, 2 * II, act16, II, (long)TT * II);
    tr(w_dm + (size_t)e * II * DD, wdT, II, DD);
    gemm(act16, wdT, ybuf, TT, DD, II);
    moe_accum<<<dim3((unsigned)(ND / 256)), dim3(256), 0, stream>>>(ybuf, comb, e, moeo);
  }

  // ---- shared MLP ----
  tr(w_gsh, wgT, DD, ISHH);
  tr(w_ush, wuT, DD, ISHH);
  tr(w_dsh, wdshT, ISHH, DD);
  gemm(xn2, wgT, gsh, TT, ISHH, DD);
  gemm(xn2, wuT, ush, TT, ISHH, DD);
  silu_mul<<<dim3((unsigned)(((long)TT * ISHH + 255) / 256)), dim3(256), 0, stream>>>(
      gsh, ISHH, ush, ISHH, actsh, ISHH, (long)TT * ISHH);
  gemm(actsh, wdshT, shout, TT, DD, ISHH);

  // ---- final residual ----
  add3<<<dim3((unsigned)(ND / 256)), dim3(256), 0, stream>>>(hbuf, moeo, shout, out);
}
